// MultiHeadAttention_6193342841116
// MI455X (gfx1250) — matrix operand order repaired and checked by reading, not compiled, not run on hardware
//
#include <hip/hip_runtime.h>
#include <hip/hip_bf16.h>
#include <math.h>

typedef __attribute__((ext_vector_type(16))) _Float16 v16h;
typedef __attribute__((ext_vector_type(8)))  float    v8f;
typedef unsigned int v4u __attribute__((ext_vector_type(4)));
typedef int          v4i __attribute__((ext_vector_type(4)));
typedef int          v8i __attribute__((ext_vector_type(8)));

#define D_MODEL 1024
#define NHEADS  16
#define DK      64
#define BATCH   2
#define SEQ     2048
#define MROWS   (BATCH * SEQ)   // 4096

#if __has_builtin(__builtin_amdgcn_tensor_load_to_lds)
#define HAVE_TDM 1
#if __has_include(<hip/amd_detail/amd_gfx1250_TDM.h>)
#define TDM_6ARG 1
#endif
#endif

// ---------------------------------------------------------------------------
// WMMA operand loaders (ISA 7.12.2 layouts, wave32)
// A 16x32 f16: lane L<16 -> M=L, elems {K0..7, K16..23}; L>=16 -> M=L-16,
//              elems {K8..15, K24..31}
// B 32x16 f16: lane L -> N=L&15, elems {K0..7, K16..23} (L<16), {K8..15, K24..31} (L>=16), as for A
// C/D 16x16 f32: vgpr r, lane L -> M = r + (L<16?0:8), N = L&15
// ---------------------------------------------------------------------------
static __device__ inline v16h wmma_a_from_f32(const float* X, int ld, int row0, int k0) {
  const int lane = threadIdx.x & 31;
  const int m    = lane & 15;
  const int kb   = k0 + ((lane >> 4) << 3);
  const float* p = X + (size_t)(row0 + m) * ld + kb;
  v16h a;
#pragma unroll
  for (int i = 0; i < 8; ++i) { a[i] = (_Float16)p[i]; a[i + 8] = (_Float16)p[i + 16]; }
  return a;
}

static __device__ inline v16h wmma_a_from_f16(const _Float16* X, int ld, int row0, int k0) {
  const int lane = threadIdx.x & 31;
  const int m    = lane & 15;
  const int kb   = k0 + ((lane >> 4) << 3);
  const _Float16* p = X + (size_t)(row0 + m) * ld + kb;
  v16h a;
#pragma unroll
  for (int i = 0; i < 8; ++i) { a[i] = p[i]; a[i + 8] = p[i + 16]; }
  return a;
}

// B[k,n] where column n is contiguous along k in memory: addr = C + (n0+n)*ld + k
static __device__ inline v16h wmma_b_cols_f32(const float* C, int ld, int n0, int k0) {
  const int lane = threadIdx.x & 31;
  const int n    = lane & 15;
  const int kb   = k0 + ((lane >> 4) << 3);
  const float* p = C + (size_t)(n0 + n) * ld + kb;
  v16h b;
#pragma unroll
  for (int i = 0; i < 8; ++i) { b[i] = (_Float16)p[i]; b[i + 8] = (_Float16)p[i + 16]; }
  return b;
}

static __device__ inline v16h wmma_b_cols_f16(const _Float16* C, int ld, int n0, int k0) {
  const int lane = threadIdx.x & 31;
  const int n    = lane & 15;
  const int kb   = k0 + ((lane >> 4) << 3);
  const _Float16* p = C + (size_t)(n0 + n) * ld + kb;
  v16h b;
#pragma unroll
  for (int i = 0; i < 8; ++i) { b[i] = p[i]; b[i + 8] = p[i + 16]; }
  return b;
}

static __device__ inline float rmax16(float v) {
  v = fmaxf(v, __shfl_xor(v, 1));
  v = fmaxf(v, __shfl_xor(v, 2));
  v = fmaxf(v, __shfl_xor(v, 4));
  v = fmaxf(v, __shfl_xor(v, 8));
  return v;
}
static __device__ inline float rsum16(float v) {
  v += __shfl_xor(v, 1);
  v += __shfl_xor(v, 2);
  v += __shfl_xor(v, 4);
  v += __shfl_xor(v, 8);
  return v;
}

// ---------------------------------------------------------------------------
// Tensor Data Mover: 2D tile (tile_d1 rows x tile_d0 elems, f16) -> LDS.
// D# packing per ISA 8.3/8.4: group0 {count=1, lds_addr, global_addr, type=2},
// group1 {data_size=2B, tensor dims/strides, tile dims}, groups 2/3 zero (2D).
// ---------------------------------------------------------------------------
#if defined(HAVE_TDM)
static __device__ inline void tdm_load_2d_f16(unsigned lds_byte_addr, const void* gptr,
                                              unsigned tensor_d0, unsigned tensor_d1,
                                              unsigned tile_d0, unsigned tile_d1,
                                              unsigned stride0) {
  const unsigned long long ga = (unsigned long long)(uintptr_t)gptr;
  v4u g0;
  g0[0] = 1u;                                   // count=1, user descriptor
  g0[1] = lds_byte_addr;                        // lds_addr
  g0[2] = (unsigned)ga;                         // global_addr[31:0]
  g0[3] = (unsigned)((ga >> 32) & 0x1FFFFFFu) | 0x80000000u;  // addr[56:32] | type=2
  v8i g1;
  g1[0] = (int)0x00010000u;                     // workgroup_mask=0, data_size=1 (2B)
  g1[1] = (int)(tensor_d0 << 16);               // tensor_dim0[15:0]
  g1[2] = (int)((tensor_d0 >> 16) | (tensor_d1 << 16));
  g1[3] = (int)((tensor_d1 >> 16) | (tile_d0 << 16));
  g1[4] = (int)tile_d1;                         // tile_dim1, tile_dim2=0
  g1[5] = (int)stride0;                         // tensor_dim0_stride[31:0]
  g1[6] = 0;
  g1[7] = 0;
  v4i z = {0, 0, 0, 0};
#if defined(TDM_6ARG)
  v8i z8 = {0, 0, 0, 0, 0, 0, 0, 0};
  __builtin_amdgcn_tensor_load_to_lds(g0, g1, z, z, z8, 0);
#else
  __builtin_amdgcn_tensor_load_to_lds(g0, g1, z, z, 0);
#endif
}

static __device__ inline void wait_tensorcnt0() {
#if __has_builtin(__builtin_amdgcn_s_wait_tensorcnt)
  __builtin_amdgcn_s_wait_tensorcnt((short)0);
#else
  asm volatile("s_wait_tensorcnt 0" ::: "memory");
#endif
}
#endif

// Low 32 bits of a generic pointer to LDS == workgroup-relative LDS byte offset
static __device__ inline unsigned lds_offset(const void* p) {
  return (unsigned)(uintptr_t)p;
}

// ---------------------------------------------------------------------------
// Projection: Y = X @ W^T + b  (X fp32 [4096,1024], W fp32 [1024,1024])
// VMODE 0: store f16 heads [B,H,S,64]   (Q, K)
// VMODE 1: store f16 transposed heads [B,H,64,S]  (V)
// One wave = one 16x16 output tile; 8 waves / block.
// ---------------------------------------------------------------------------
template <int VMODE>
__global__ __launch_bounds__(256)
void mha_proj_qkv(const float* __restrict__ X, const float* __restrict__ W,
                  const float* __restrict__ bias, _Float16* __restrict__ dst) {
  const int w  = blockIdx.x * 8 + (threadIdx.x >> 5);
  const int m0 = (w >> 6) << 4;   // 256 M-tiles
  const int n0 = (w & 63) << 4;   // 64  N-tiles
  v8f c = {};
  for (int k = 0; k < D_MODEL; k += 32) {
    v16h a = wmma_a_from_f32(X, D_MODEL, m0, k);
    v16h b = wmma_b_cols_f32(W, D_MODEL, n0, k);   // B[k,n] = W[n,k]
    c = __builtin_amdgcn_wmma_f32_16x16x32_f16(false, a, false, b, (short)0, c, false, false);
  }
  const int lane = threadIdx.x & 31;
  const int n    = n0 + (lane & 15);
  const int h    = n >> 6, d = n & 63;
  const float bn = bias[n];
#pragma unroll
  for (int r = 0; r < 8; ++r) {
    const int m  = m0 + r + ((lane >> 4) << 3);
    const int bb = m >> 11;        // batch
    const int s  = m & 2047;       // sequence pos
    const float y = c[r] + bn;
    if (VMODE == 0)
      dst[((((size_t)bb * NHEADS + h) * SEQ + s) << 6) + d] = (_Float16)y;
    else
      dst[(((size_t)bb * NHEADS + h) * DK + d) * SEQ + s] = (_Float16)y;
  }
}

// ---------------------------------------------------------------------------
// Flash attention. Block = 8 waves, all on the SAME (batch,head): blockIdx ->
// (bh, query-group of 8x16 rows). Each key chunk (32 keys) is staged once per
// block into LDS (TDM by wave 0 when available, cooperative copy otherwise)
// and consumed by all 8 waves.
// Qh,Kh: [B,H,S,64] f16; Vt: [B,H,64,S] f16; O: [B,S,1024] f16
// ---------------------------------------------------------------------------
__global__ __launch_bounds__(256)
void mha_attention(const _Float16* __restrict__ Qh, const _Float16* __restrict__ Kh,
                   const _Float16* __restrict__ Vt, _Float16* __restrict__ O) {
  __shared__ __align__(16) _Float16 lds_k[32][DK];      // 32 keys x 64   (4 KB)
  __shared__ __align__(16) _Float16 lds_v[DK][32];      // 64 x 32 keys   (4 KB)
  __shared__ __align__(16) _Float16 lds_p[8][16][32];   // per-wave P-tile (8 KB)

  const int widx = threadIdx.x >> 5;
  const int lane = threadIdx.x & 31;
  const int bh   = blockIdx.x >> 4;               // 0..31 (b*16+h), block-uniform
  const int q0   = (((blockIdx.x & 15) << 3) + widx) << 4;  // query tile base

  const _Float16* Qbase = Qh + (size_t)bh * SEQ * DK;   // [2048][64]
  const _Float16* Kbase = Kh + (size_t)bh * SEQ * DK;   // [2048][64]
  const _Float16* Vbase = Vt + (size_t)bh * DK * SEQ;   // [64][2048]

  // Q tile, 16x64, as two A operands (K = 0..31, 32..63)
  const v16h aq0 = wmma_a_from_f16(Qbase, DK, q0, 0);
  const v16h aq1 = wmma_a_from_f16(Qbase, DK, q0, 32);

  v8f acc0 = {}, acc1 = {}, acc2 = {}, acc3 = {};   // 16x64 f32 accumulator
  float mrow[8], lrow[8];
#pragma unroll
  for (int r = 0; r < 8; ++r) { mrow[r] = -INFINITY; lrow[r] = 0.f; }

  const int rowoff = (lane >> 4) << 3;
  const int nlane  = lane & 15;

  for (int kt = 0; kt < SEQ; kt += 32) {
    // ---- stage shared K/V chunk into LDS --------------------------------
    __syncthreads();                               // previous chunk fully read
#if defined(HAVE_TDM)
    if (widx == 0) {
      // K tile: 32 rows x 64 f16, row stride 64 (contiguous block)
      tdm_load_2d_f16(lds_offset(&lds_k[0][0]), Kbase + (size_t)kt * DK,
                      /*tensor_d0=*/DK, /*tensor_d1=*/SEQ,
                      /*tile_d0=*/DK, /*tile_d1=*/32, /*stride0=*/DK);
      // V tile: 64 rows x 32 f16 out of [64][2048]
      tdm_load_2d_f16(lds_offset(&lds_v[0][0]), Vbase + kt,
                      /*tensor_d0=*/SEQ, /*tensor_d1=*/DK,
                      /*tile_d0=*/32, /*tile_d1=*/DK, /*stride0=*/SEQ);
      wait_tensorcnt0();
    }
#else
    {
      // K: 4 KB contiguous; 256 threads x 16 B
      const uint4* ksrc = (const uint4*)(Kbase + (size_t)kt * DK);
      ((uint4*)&lds_k[0][0])[threadIdx.x] = ksrc[threadIdx.x];
      // V: 64 rows x 32 f16 (64 B per row); 4 threads per row
      const int vr = threadIdx.x >> 2;
      const int vc = (threadIdx.x & 3) << 3;
      *(uint4*)(&lds_v[vr][vc]) = *(const uint4*)(Vbase + (size_t)vr * SEQ + kt + vc);
    }
#endif
    __syncthreads();                               // tiles visible to all waves

    // ---- scores: 16 queries x 32 keys (two 16-wide N slices) ------------
    v8f s0 = {}, s1 = {};
    {
      v16h bk;
      bk = wmma_b_cols_f16(&lds_k[0][0], DK, 0, 0);    // B[kd, key]
      s0 = __builtin_amdgcn_wmma_f32_16x16x32_f16(false, aq0, false, bk, (short)0, s0, false, false);
      bk = wmma_b_cols_f16(&lds_k[0][0], DK, 0, 32);
      s0 = __builtin_amdgcn_wmma_f32_16x16x32_f16(false, aq1, false, bk, (short)0, s0, false, false);
      bk = wmma_b_cols_f16(&lds_k[0][0], DK, 16, 0);
      s1 = __builtin_amdgcn_wmma_f32_16x16x32_f16(false, aq0, false, bk, (short)0, s1, false, false);
      bk = wmma_b_cols_f16(&lds_k[0][0], DK, 16, 32);
      s1 = __builtin_amdgcn_wmma_f32_16x16x32_f16(false, aq1, false, bk, (short)0, s1, false, false);
    }
    // ---- online softmax update ------------------------------------------
#pragma unroll
    for (int r = 0; r < 8; ++r) {
      const float e0 = s0[r] * 0.125f;               // 1/sqrt(64)
      const float e1 = s1[r] * 0.125f;
      const float cm = rmax16(fmaxf(e0, e1));
      const float mn = fmaxf(mrow[r], cm);
      const float alpha = __expf(mrow[r] - mn);
      const float p0 = __expf(e0 - mn);
      const float p1 = __expf(e1 - mn);
      mrow[r] = mn;
      lrow[r] = lrow[r] * alpha + rsum16(p0 + p1);
      acc0[r] *= alpha; acc1[r] *= alpha; acc2[r] *= alpha; acc3[r] *= alpha;
      lds_p[widx][r + rowoff][nlane]      = (_Float16)p0;
      lds_p[widx][r + rowoff][16 + nlane] = (_Float16)p1;
    }
    asm volatile("s_wait_dscnt 0" ::: "memory");     // C-layout -> A-layout bounce
    // ---- reload P (16x32) in A-operand layout ---------------------------
    v16h ap;
    {
      const int kb = (lane >> 4) << 3;
#pragma unroll
      for (int i = 0; i < 8; ++i) {
        ap[i]     = lds_p[widx][nlane][kb + i];
        ap[i + 8] = lds_p[widx][nlane][kb + 16 + i];
      }
    }
    // ---- acc += P @ V  (B[key, d] = lds_v[d][key], contig along key) ----
    v16h bv;
    bv   = wmma_b_cols_f16(&lds_v[0][0], 32, 0, 0);
    acc0 = __builtin_amdgcn_wmma_f32_16x16x32_f16(false, ap, false, bv, (short)0, acc0, false, false);
    bv   = wmma_b_cols_f16(&lds_v[0][0], 32, 16, 0);
    acc1 = __builtin_amdgcn_wmma_f32_16x16x32_f16(false, ap, false, bv, (short)0, acc1, false, false);
    bv   = wmma_b_cols_f16(&lds_v[0][0], 32, 32, 0);
    acc2 = __builtin_amdgcn_wmma_f32_16x16x32_f16(false, ap, false, bv, (short)0, acc2, false, false);
    bv   = wmma_b_cols_f16(&lds_v[0][0], 32, 48, 0);
    acc3 = __builtin_amdgcn_wmma_f32_16x16x32_f16(false, ap, false, bv, (short)0, acc3, false, false);
  }

  // ---- normalize and store as [B,S,H*64] f16 (ready for output GEMM) ----
  const int bb = bh >> 4, h = bh & 15;
  const int c0 = h << 6;
#pragma unroll
  for (int r = 0; r < 8; ++r) {
    const int s = q0 + r + rowoff;
    const float inv = 1.0f / lrow[r];
    const size_t base = ((size_t)bb * SEQ + s) * D_MODEL + c0 + nlane;
    O[base]      = (_Float16)(acc0[r] * inv);
    O[base + 16] = (_Float16)(acc1[r] * inv);
    O[base + 32] = (_Float16)(acc2[r] * inv);
    O[base + 48] = (_Float16)(acc3[r] * inv);
  }
}

// ---------------------------------------------------------------------------
// Output projection: out = A @ Wo^T + bo  (A f16 [4096,1024] -> fp32 out)
// ---------------------------------------------------------------------------
__global__ __launch_bounds__(256)
void mha_out_proj(const _Float16* __restrict__ A, const float* __restrict__ W,
                  const float* __restrict__ bias, float* __restrict__ Y) {
  const int w  = blockIdx.x * 8 + (threadIdx.x >> 5);
  const int m0 = (w >> 6) << 4;
  const int n0 = (w & 63) << 4;
  v8f c = {};
  for (int k = 0; k < D_MODEL; k += 32) {
    v16h a = wmma_a_from_f16(A, D_MODEL, m0, k);
    v16h b = wmma_b_cols_f32(W, D_MODEL, n0, k);
    c = __builtin_amdgcn_wmma_f32_16x16x32_f16(false, a, false, b, (short)0, c, false, false);
  }
  const int lane = threadIdx.x & 31;
  const int n    = n0 + (lane & 15);
  const float bn = bias[n];
#pragma unroll
  for (int r = 0; r < 8; ++r) {
    const int m = m0 + r + ((lane >> 4) << 3);
    Y[(size_t)m * D_MODEL + n] = c[r] + bn;
  }
}

// ---------------------------------------------------------------------------
extern "C" void kernel_launch(void* const* d_in, const int* in_sizes, int n_in,
                              void* d_out, int out_size, void* d_ws, size_t ws_size,
                              hipStream_t stream) {
  const float* q  = (const float*)d_in[0];
  const float* k  = (const float*)d_in[1];
  const float* v  = (const float*)d_in[2];
  const float* Wq = (const float*)d_in[3];
  const float* bq = (const float*)d_in[4];
  const float* Wk = (const float*)d_in[5];
  const float* bk = (const float*)d_in[6];
  const float* Wv = (const float*)d_in[7];
  const float* bv = (const float*)d_in[8];
  const float* Wo = (const float*)d_in[9];
  const float* bo = (const float*)d_in[10];
  float* out = (float*)d_out;

  // workspace: 4 x (4096*1024) f16 buffers = 32 MB
  _Float16* Qh = (_Float16*)d_ws;
  _Float16* Kh = Qh + (size_t)MROWS * D_MODEL;
  _Float16* Vt = Kh + (size_t)MROWS * D_MODEL;
  _Float16* AO = Vt + (size_t)MROWS * D_MODEL;

  const dim3 blk(256);
  const int gemm_blocks = (MROWS / 16) * (D_MODEL / 16) / 8;          // 2048
  const int attn_blocks = BATCH * NHEADS * (SEQ / 16) / 8;            // 512

  hipLaunchKernelGGL((mha_proj_qkv<0>), dim3(gemm_blocks), blk, 0, stream, q, Wq, bq, Qh);
  hipLaunchKernelGGL((mha_proj_qkv<0>), dim3(gemm_blocks), blk, 0, stream, k, Wk, bk, Kh);
  hipLaunchKernelGGL((mha_proj_qkv<1>), dim3(gemm_blocks), blk, 0, stream, v, Wv, bv, Vt);
  hipLaunchKernelGGL(mha_attention, dim3(attn_blocks), blk, 0, stream, Qh, Kh, Vt, AO);
  hipLaunchKernelGGL(mha_out_proj,  dim3(gemm_blocks), blk, 0, stream, AO, Wo, bo, out);
}
